// WavLM_9543417332329
// MI455X (gfx1250) — matrix operand order repaired and checked by reading, not compiled, not run on hardware
//
#include <hip/hip_runtime.h>

typedef unsigned short ushort_t;
typedef unsigned int uint32;
typedef __attribute__((ext_vector_type(16))) __bf16 v16bf;
typedef __attribute__((ext_vector_type(8)))  float  v8f;

struct alignas(16) U4 { uint32 x, y, z, w; };
union Frag { U4 q[2]; v16bf v; };

__device__ __forceinline__ float gelu_f(float x) {
    return 0.5f * x * (1.0f + erff(x * 0.70710678118654752f));
}
__device__ __forceinline__ ushort_t f2bf(float f) {
    uint32 u = __float_as_uint(f);
    uint32 r = (u + 0x7fffu + ((u >> 16) & 1u)) >> 16;
    return (ushort_t)r;
}
__device__ __forceinline__ float bf2f(ushort_t h) {
    return __uint_as_float(((uint32)h) << 16);
}

// ---------------------------------------------------------------------------
// Generic batched BF16 WMMA GEMM:  C[z] = act((A[z] * B[z]^T + bias) * alpha)
// A: [M,K] rows strided lda ; B: [N,K] rows strided ldb (i.e. C=A*B^T)
// batch z decomposed as z1=z/bdiv, z2=z%bdiv with independent strides.
// Block: 256 threads = 8 waves; block tile 128x128; K-step 64.
// Each wave: 32 rows x 64 cols = 2x4 WMMA tiles -> 16 v_wmma per K-stage.
// Out-of-range rows are CLAMPED (not zero-filled): garbage stays confined to
// out-of-range output rows/cols, which are masked at store.
// Requires: K % 64 == 0 (true for every GEMM in this model).
// flags: 1 = add bias[col], 2 = GELU
// ---------------------------------------------------------------------------
__global__ __launch_bounds__(256)
void gemm_wmma(const ushort_t* __restrict__ A, long lda, long sA1, long sA2,
               const ushort_t* __restrict__ Bw, long ldb, long sB1, long sB2,
               const float* __restrict__ bias,
               float* __restrict__ Cf, ushort_t* __restrict__ Cb,
               long ldc, long sC1, long sC2,
               int M, int N, int K, float alpha, int bdiv, int flags)
{
    int z  = blockIdx.z;
    int z1 = z / bdiv, z2 = z % bdiv;
    A  += z1 * sA1 + z2 * sA2;
    Bw += z1 * sB1 + z2 * sB2;
    long coff = (long)z1 * sC1 + (long)z2 * sC2;

    int m0 = blockIdx.y * 128;
    int n0 = blockIdx.x * 128;

    __shared__ U4 As[128][8];   // 128 rows x 64 bf16
    __shared__ U4 Bs[128][8];

    int tid  = threadIdx.x;
    int lane = tid & 31;
    int w    = tid >> 5;
    int wr   = (w >> 1) * 32;   // wave row offset in 128x128 tile (4 row groups)
    int wc   = (w & 1) * 64;    // wave col offset (2 col groups)

    int half = (lane >> 4) & 1;
    int rsel = lane & 15;

    v8f acc[2][4] = {};

    for (int k0 = 0; k0 < K; k0 += 64) {
        // stage A & B tiles: 4 chunks each per thread, unconditional clamped loads
#pragma unroll
        for (int c = tid; c < 1024; c += 256) {
            int row = c >> 3, cc = c & 7;
            int ra = m0 + row; ra = (ra < M) ? ra : (M - 1);
            As[row][cc] = *(const U4*)(A + (long)ra * lda + k0 + cc * 8);
            int rb = n0 + row; rb = (rb < N) ? rb : (N - 1);
            Bs[row][cc] = *(const U4*)(Bw + (long)rb * ldb + k0 + cc * 8);
        }
        __syncthreads();

#pragma unroll
        for (int ks = 0; ks < 2; ++ks) {
            Frag fa[2], fb[4];
#pragma unroll
            for (int i = 0; i < 2; ++i) {
                int ar = wr + i * 16 + rsel;
                fa[i].q[0] = As[ar][ks * 4 + half];        // A: K 0-7 / 8-15
                fa[i].q[1] = As[ar][ks * 4 + half + 2];    //    K 16-23 / 24-31
            }
#pragma unroll
            for (int j = 0; j < 4; ++j) {
                int br = wc + j * 16 + rsel;
                fb[j].q[0] = Bs[br][ks * 4 + half];        // B: K 0-7 / 8-15
                fb[j].q[1] = Bs[br][ks * 4 + half + 2];    //    K 16-23 / 24-31
            }
#pragma unroll
            for (int i = 0; i < 2; ++i)
#pragma unroll
                for (int j = 0; j < 4; ++j)
                    acc[i][j] = __builtin_amdgcn_wmma_f32_16x16x32_bf16(
                        false, fa[i].v, false, fb[j].v, (short)0, acc[i][j], false, false);
        }
        __syncthreads();
    }

    // store: VGPR r -> M = r + (lane>=16 ? 8 : 0), N = lane & 15
    int cn    = lane & 15;
    int rbase = (lane >> 4) * 8;
#pragma unroll
    for (int i = 0; i < 2; ++i)
#pragma unroll
        for (int j = 0; j < 4; ++j) {
            int col = n0 + wc + j * 16 + cn;
            if (col >= N) continue;
            float bval = (flags & 1) ? bias[col] : 0.0f;
#pragma unroll
            for (int r = 0; r < 8; ++r) {
                int row = m0 + wr + i * 16 + rbase + r;
                if (row >= M) continue;
                float v = (acc[i][j][r] + bval) * alpha;
                if (flags & 2) v = gelu_f(v);
                long idx = coff + (long)row * ldc + col;
                if (Cf) Cf[idx] = v;
                if (Cb) Cb[idx] = f2bf(v);
            }
        }
}

// ---------------------------------------------------------------------------
// conv0 (1->512, k=10, s=5) fused with channel-LN + GELU, bf16 channels-last out
// ---------------------------------------------------------------------------
__global__ __launch_bounds__(256)
void conv0_ln_gelu(const float* __restrict__ src, const float* __restrict__ w0,
                   const float* __restrict__ g, const float* __restrict__ b,
                   ushort_t* __restrict__ X0, int T, int T0)
{
    long blk = blockIdx.x;            // b*T0 + t
    long bb = blk / T0;
    int  t  = (int)(blk % T0);
    __shared__ float win[10];
    __shared__ float red[256];
    int tid = threadIdx.x;
    if (tid < 10) win[tid] = src[bb * T + (long)t * 5 + tid];
    __syncthreads();
    float o[2];
    for (int i = 0; i < 2; ++i) {
        int c = tid + i * 256;
        float s = 0.0f;
        for (int k = 0; k < 10; ++k) s += w0[c * 10 + k] * win[k];
        o[i] = s;
    }
    red[tid] = o[0] + o[1]; __syncthreads();
    for (int off = 128; off > 0; off >>= 1) { if (tid < off) red[tid] += red[tid + off]; __syncthreads(); }
    float mean = red[0] / 512.0f; __syncthreads();
    float d0 = o[0] - mean, d1 = o[1] - mean;
    red[tid] = d0 * d0 + d1 * d1; __syncthreads();
    for (int off = 128; off > 0; off >>= 1) { if (tid < off) red[tid] += red[tid + off]; __syncthreads(); }
    float rstd = rsqrtf(red[0] / 512.0f + 1e-5f);
    for (int i = 0; i < 2; ++i) {
        int c = tid + i * 256;
        float y = (o[i] - mean) * rstd * g[c] + b[c];
        X0[blk * 512 + c] = f2bf(gelu_f(y));
    }
}

// ---------------------------------------------------------------------------
// Row LayerNorm (+optional residual, +optional GELU), writes fp32 and/or bf16
// ---------------------------------------------------------------------------
__global__ __launch_bounds__(256)
void ln_rows(const float* __restrict__ X, const float* __restrict__ R,
             const float* __restrict__ g, const float* __restrict__ bta,
             float* __restrict__ outf, ushort_t* __restrict__ outb,
             int C, int gelu_flag)
{
    long row = blockIdx.x;
    const float* x = X + row * C;
    const float* r = R ? R + row * C : nullptr;
    int tid = threadIdx.x;
    int nv = (C + 255) / 256;     // <= 3 for C<=768
    float v[3];
    float s = 0.0f;
    for (int i = 0; i < nv; ++i) {
        int c = tid + i * 256;
        float val = 0.0f;
        if (c < C) { val = x[c]; if (r) val += r[c]; }
        v[i] = val; s += val;
    }
    __shared__ float red[256];
    red[tid] = s; __syncthreads();
    for (int off = 128; off > 0; off >>= 1) { if (tid < off) red[tid] += red[tid + off]; __syncthreads(); }
    float mean = red[0] / C; __syncthreads();
    float sq = 0.0f;
    for (int i = 0; i < nv; ++i) {
        int c = tid + i * 256;
        if (c < C) { float d = v[i] - mean; sq += d * d; }
    }
    red[tid] = sq; __syncthreads();
    for (int off = 128; off > 0; off >>= 1) { if (tid < off) red[tid] += red[tid + off]; __syncthreads(); }
    float rstd = rsqrtf(red[0] / C + 1e-5f);
    for (int i = 0; i < nv; ++i) {
        int c = tid + i * 256;
        if (c < C) {
            float y = (v[i] - mean) * rstd * g[c] + bta[c];
            if (gelu_flag) y = gelu_f(y);
            if (outf) outf[row * C + c] = y;
            if (outb) outb[row * C + c] = f2bf(y);
        }
    }
}

// im2col for conv layers (channels-last bf16 in, A[M, 512*kw] out, K=ci*kw+kk)
__global__ void im2col_conv(const ushort_t* __restrict__ Xin, ushort_t* __restrict__ A,
                            int Tin, int Tout, int kw, int stride, long total, int K)
{
    long idx = (long)blockIdx.x * 256 + threadIdx.x;
    if (idx >= total) return;
    long m = idx / K;
    int kidx = (int)(idx % K);
    int ci = kidx / kw, kk = kidx - ci * kw;
    long b = m / Tout;
    int  t = (int)(m % Tout);
    A[idx] = Xin[((b * Tin) + (long)t * stride + kk) * 512 + ci];
}

// im2col for grouped positional conv (16 groups, K = ci_local*128 + kk, pad 64)
__global__ void im2col_posconv(const ushort_t* __restrict__ Xb, ushort_t* __restrict__ A, long total)
{
    long idx = (long)blockIdx.x * 256 + threadIdx.x;
    if (idx >= total) return;
    const long perG = 1996L * 6144;
    int g = (int)(idx / perG);
    long rem = idx % perG;
    long m = rem / 6144;
    int kidx = (int)(rem % 6144);
    int ci = kidx >> 7, kk = kidx & 127;
    int b = (int)(m / 499), t = (int)(m % 499);
    int tt = t + kk - 64;
    ushort_t val = 0;
    if (tt >= 0 && tt < 499)
        val = Xb[((long)b * 499 + tt) * 768 + g * 48 + ci];
    A[idx] = val;
}

// xc = gelu(posout + bias[col]); out = xproj + xc
__global__ void posconv_add(const float* __restrict__ xproj, const float* __restrict__ posout,
                            const float* __restrict__ pb, float* __restrict__ out, long total)
{
    long idx = (long)blockIdx.x * 256 + threadIdx.x;
    if (idx >= total) return;
    int col = (int)(idx % 768);
    out[idx] = xproj[idx] + gelu_f(posout[idx] + pb[col]);
}

// T5 bucketization -> pos_bias[h][s][t]
__global__ void posbias_kernel(const float* __restrict__ rel, float* __restrict__ pb, int S)
{
    int idx = blockIdx.x * 256 + threadIdx.x;
    if (idx >= S * S) return;
    int s = idx / S, t = idx % S;
    int r = t - s;
    int b0 = (r > 0) ? 160 : 0;
    int n = (r < 0) ? -r : r;
    int loc;
    if (n < 80) loc = n;
    else {
        int large = 80 + (int)(logf((float)n / 80.0f) * (80.0f / logf(10.0f)));
        loc = (large < 159) ? large : 159;
    }
    int bucket = b0 + loc;
    for (int h = 0; h < 12; ++h)
        pb[((long)h * S + s) * S + t] = rel[bucket * 12 + h];
}

// gate[b,h,s] = g0*(g1*a[h]-1)+2 with g0/g1 sums of 4 sigmoids of 8 dots
__global__ void gate_kernel(const float* __restrict__ xf, const float* __restrict__ gw,
                            const float* __restrict__ gb, const float* __restrict__ ga,
                            float* __restrict__ gate, int S, int total)
{
    int idx = blockIdx.x * 256 + threadIdx.x;
    if (idx >= total) return;
    int s = idx % S;
    int h = (idx / S) % 12;
    int b = idx / (S * 12);
    const float* q = xf + ((long)(b * S + s)) * 768 + h * 64;
    float sg[8];
    for (int j = 0; j < 8; ++j) {
        float acc = gb[j];
        const float* w = gw + j * 64;
        for (int k = 0; k < 64; ++k) acc += q[k] * w[k];
        sg[j] = 1.0f / (1.0f + expf(-acc));
    }
    float g0 = sg[0] + sg[1] + sg[2] + sg[3];
    float g1 = sg[4] + sg[5] + sg[6] + sg[7];
    gate[idx] = g0 * (g1 * ga[h] - 1.0f) + 2.0f;
}

// softmax over t of scores + gate*pos_bias; bf16 probs with zero pad to Sp
__global__ __launch_bounds__(256)
void softmax_kernel(const float* __restrict__ scores, const float* __restrict__ gate,
                    const float* __restrict__ pb, ushort_t* __restrict__ probs,
                    int S, int Sp)
{
    long row = blockIdx.x;                  // (b*12+h)*S + s
    int s = (int)(row % S);
    int h = (int)((row / S) % 12);
    const float* sc = scores + row * Sp;
    float gv = gate[row];
    const float* pbr = pb + ((long)h * S + s) * S;
    __shared__ float red[256];
    int tid = threadIdx.x;
    float v[2];
    float mx = -1e30f;
    for (int i = 0; i < 2; ++i) {
        int t = tid + i * 256;
        float val = -1e30f;
        if (t < S) val = sc[t] + gv * pbr[t];
        v[i] = val;
        mx = fmaxf(mx, val);
    }
    red[tid] = mx; __syncthreads();
    for (int off = 128; off > 0; off >>= 1) { if (tid < off) red[tid] = fmaxf(red[tid], red[tid + off]); __syncthreads(); }
    mx = red[0]; __syncthreads();
    float sum = 0.0f;
    for (int i = 0; i < 2; ++i) {
        int t = tid + i * 256;
        float e = 0.0f;
        if (t < S) e = expf(v[i] - mx);
        v[i] = e; sum += e;
    }
    red[tid] = sum; __syncthreads();
    for (int off = 128; off > 0; off >>= 1) { if (tid < off) red[tid] += red[tid + off]; __syncthreads(); }
    float inv = 1.0f / red[0];
    for (int i = 0; i < 2; ++i) {
        int t = tid + i * 256;
        if (t < Sp) probs[row * Sp + t] = f2bf((t < S) ? v[i] * inv : 0.0f);
    }
}

// vt[(b*12+h)*64+d][t(512 pad)] = V[(b*499+t)*768 + h*64 + d]
__global__ void vtrans_kernel(const ushort_t* __restrict__ vbuf, ushort_t* __restrict__ vt, long total)
{
    long idx = (long)blockIdx.x * 256 + threadIdx.x;
    if (idx >= total) return;
    int t  = (int)(idx & 511);
    int d  = (int)((idx >> 9) & 63);
    int bh = (int)(idx >> 15);
    int b = bh / 12, h = bh % 12;
    ushort_t val = 0;
    if (t < 499) val = vbuf[((long)b * 499 + t) * 768 + h * 64 + d];
    vt[idx] = val;
}

__global__ void f2bf_kernel(const float* __restrict__ src, ushort_t* __restrict__ dst, long n)
{
    long i = (long)blockIdx.x * 256 + threadIdx.x;
    if (i < n) dst[i] = f2bf(src[i]);
}

// ---------------------------------------------------------------------------
extern "C" void kernel_launch(void* const* d_in, const int* in_sizes, int n_in,
                              void* d_out, int out_size, void* d_ws, size_t ws_size,
                              hipStream_t stream)
{
    (void)in_sizes; (void)n_in; (void)out_size; (void)ws_size;
    auto IN = [&](int i) { return (const float*)d_in[i]; };

    char* base = (char*)d_ws;
    size_t off = 0;
    auto alloc = [&](size_t bytes) -> char* {
        char* p = base + off;
        off += (bytes + 255) & ~(size_t)255;
        return p;
    };

    const int S = 499;
    const long MS = 1996;         // B*S

    // persistent bf16 weights
    ushort_t* wqb  = (ushort_t*)alloc(12ll * 768 * 768 * 2);
    ushort_t* wkb  = (ushort_t*)alloc(12ll * 768 * 768 * 2);
    ushort_t* wvb  = (ushort_t*)alloc(12ll * 768 * 768 * 2);
    ushort_t* wob  = (ushort_t*)alloc(12ll * 768 * 768 * 2);
    ushort_t* f1b  = (ushort_t*)alloc(12ll * 3072 * 768 * 2);
    ushort_t* f2wb = (ushort_t*)alloc(12ll * 3072 * 768 * 2);
    ushort_t* c14b = (ushort_t*)alloc(4ll * 512 * 1536 * 2);
    ushort_t* c56b = (ushort_t*)alloc(2ll * 512 * 1024 * 2);
    ushort_t* projbw = (ushort_t*)alloc(768ll * 512 * 2);
    ushort_t* poswb  = (ushort_t*)alloc(768ll * 6144 * 2);

    float*    posbias = (float*)alloc(12ll * S * S * 4);
    float*    gateb   = (float*)alloc(48ll * S * 4);
    float*    xf      = (float*)alloc(MS * 768 * 4);
    ushort_t* xb      = (ushort_t*)alloc(MS * 768 * 2);
    ushort_t* qb      = (ushort_t*)alloc(MS * 768 * 2);
    ushort_t* kbuf    = (ushort_t*)alloc(MS * 768 * 2);
    ushort_t* vbuf    = (ushort_t*)alloc(MS * 768 * 2);
    ushort_t* vt      = (ushort_t*)alloc(48ll * 64 * 512 * 2);
    float*    scores  = (float*)alloc(48ll * S * 512 * 4);
    ushort_t* probs   = (ushort_t*)alloc(48ll * S * 512 * 2);
    ushort_t* ctxb    = (ushort_t*)alloc(MS * 768 * 2);
    float*    attnf   = (float*)alloc(MS * 768 * 4);
    ushort_t* hb      = (ushort_t*)alloc(MS * 3072 * 2);
    float*    ff2f    = (float*)alloc(MS * 768 * 4);
    float*    x6f     = (float*)alloc(MS * 512 * 4);
    ushort_t* x6lnb   = (ushort_t*)alloc(MS * 512 * 2);
    float*    xprojf  = (float*)alloc(MS * 768 * 4);
    ushort_t* xprojb  = (ushort_t*)alloc(MS * 768 * 2);
    float*    posoutf = (float*)alloc(MS * 768 * 4);
    float*    posaddf = (float*)alloc(MS * 768 * 4);

    // arena (conv scratch, reused for posconv im2col)
    const size_t szXA  = 4ll * 31999 * 512 * 2;
    const size_t szXB  = 4ll * 15999 * 512 * 2;
    const size_t szAim = 63996ll * 1536 * 2;
    const size_t szG   = 63996ll * 512 * 4;
    char* arena = alloc(szXA + szXB + szAim + szG);
    ushort_t* XA   = (ushort_t*)arena;
    ushort_t* XB   = (ushort_t*)(arena + szXA);
    ushort_t* Aim  = (ushort_t*)(arena + szXA + szXB);
    float*    Gout = (float*)(arena + szXA + szXB + szAim);
    ushort_t* Apos = (ushort_t*)arena;      // overlays conv scratch (conv done by then)

    auto cvt = [&](const float* s, ushort_t* d, long n) {
        f2bf_kernel<<<(unsigned)((n + 255) / 256), 256, 0, stream>>>(s, d, n);
    };
    auto gemm = [&](const ushort_t* A, long lda, long sA1, long sA2,
                    const ushort_t* Bw, long ldb, long sB1, long sB2,
                    const float* bias, float* Cf, ushort_t* Cb,
                    long ldc, long sC1, long sC2,
                    int M, int N, int K, float alpha, int bdiv, int flags, int batch) {
        dim3 grid((N + 127) / 128, (M + 127) / 128, batch);
        gemm_wmma<<<grid, 256, 0, stream>>>(A, lda, sA1, sA2, Bw, ldb, sB1, sB2,
                                            bias, Cf, Cb, ldc, sC1, sC2,
                                            M, N, K, alpha, bdiv, flags);
    };

    // ---- weight conversions to bf16
    cvt(IN(14), wqb,  12ll * 768 * 768);
    cvt(IN(16), wkb,  12ll * 768 * 768);
    cvt(IN(18), wvb,  12ll * 768 * 768);
    cvt(IN(20), wob,  12ll * 768 * 768);
    cvt(IN(27), f1b,  12ll * 3072 * 768);
    cvt(IN(29), f2wb, 12ll * 3072 * 768);
    cvt(IN(2),  c14b, 4ll * 512 * 1536);
    cvt(IN(3),  c56b, 2ll * 512 * 1024);
    cvt(IN(8),  projbw, 768ll * 512);
    cvt(IN(10), poswb,  768ll * 6144);

    // ---- relative-position bias table
    posbias_kernel<<<(S * S + 255) / 256, 256, 0, stream>>>(IN(33), posbias, S);

    // ---- conv0 fused LN+GELU -> XA (channels-last bf16)
    conv0_ln_gelu<<<4 * 31999, 256, 0, stream>>>(IN(0), IN(1), IN(4), IN(5), XA, 160000, 31999);

    // ---- conv layers 1..6 as im2col + WMMA GEMM + LN/GELU
    int Tprev = 31999;
    ushort_t* Xprev = XA;
    for (int i = 1; i < 7; ++i) {
        int kw = (i <= 4) ? 3 : 2;
        int Tout = (Tprev - kw) / 2 + 1;
        int K = 512 * kw;
        long M = 4L * Tout;
        long total = M * K;
        im2col_conv<<<(unsigned)((total + 255) / 256), 256, 0, stream>>>(
            Xprev, Aim, Tprev, Tout, kw, 2, total, K);
        const ushort_t* wbf = (i <= 4) ? (c14b + (size_t)(i - 1) * 512 * 1536)
                                       : (c56b + (size_t)(i - 5) * 512 * 1024);
        gemm(Aim, K, 0, 0, wbf, K, 0, 0, nullptr, Gout, nullptr, 512, 0, 0,
             (int)M, 512, K, 1.0f, 1, 0, 1);
        ushort_t* Xnext = (i & 1) ? XB : XA;
        ln_rows<<<(unsigned)M, 256, 0, stream>>>(Gout, nullptr, IN(4) + i * 512, IN(5) + i * 512,
                                                 (i == 6) ? x6f : nullptr, Xnext, 512, 1);
        Xprev = Xnext; Tprev = Tout;
    }

    // ---- feature LN + projection to 768
    ln_rows<<<(unsigned)MS, 256, 0, stream>>>(x6f, nullptr, IN(6), IN(7), nullptr, x6lnb, 512, 0);
    gemm(x6lnb, 512, 0, 0, projbw, 512, 0, 0, IN(9), xprojf, xprojb, 768, 0, 0,
         (int)MS, 768, 512, 1.0f, 1, 1, 1);

    // ---- positional conv (16-group) as batched GEMM
    {
        long total = 16ll * MS * 6144;
        im2col_posconv<<<(unsigned)((total + 255) / 256), 256, 0, stream>>>(xprojb, Apos, total);
        gemm(Apos, 6144, 0, MS * 6144, poswb, 6144, 0, 48ll * 6144, nullptr,
             posoutf, nullptr, 768, 0, 48, (int)MS, 48, 6144, 1.0f, 16, 0, 16);
        posconv_add<<<(unsigned)((MS * 768 + 255) / 256), 256, 0, stream>>>(
            xprojf, posoutf, IN(11), posaddf, MS * 768);
        ln_rows<<<(unsigned)MS, 256, 0, stream>>>(posaddf, nullptr, IN(12), IN(13), xf, xb, 768, 0);
    }

    // ---- transformer layers
    const float scaling = 0.125f;   // 64^-0.5
    for (int l = 0; l < 12; ++l) {
        const ushort_t* wq = wqb  + (size_t)l * 768 * 768;
        const ushort_t* wk = wkb  + (size_t)l * 768 * 768;
        const ushort_t* wv = wvb  + (size_t)l * 768 * 768;
        const ushort_t* wo = wob  + (size_t)l * 768 * 768;
        const ushort_t* w1 = f1b  + (size_t)l * 3072 * 768;
        const ushort_t* w2 = f2wb + (size_t)l * 3072 * 768;

        // gates from raw layer input
        gate_kernel<<<(48 * S + 255) / 256, 256, 0, stream>>>(
            xf, IN(22) + l * 512, IN(23) + l * 8, IN(24) + l * 12, gateb, S, 48 * S);

        // Q/K/V projections (bf16 out; Q pre-scaled)
        gemm(xb, 768, 0, 0, wq, 768, 0, 0, IN(15) + l * 768, nullptr, qb, 768, 0, 0,
             (int)MS, 768, 768, scaling, 1, 1, 1);
        gemm(xb, 768, 0, 0, wk, 768, 0, 0, IN(17) + l * 768, nullptr, kbuf, 768, 0, 0,
             (int)MS, 768, 768, 1.0f, 1, 1, 1);
        gemm(xb, 768, 0, 0, wv, 768, 0, 0, IN(19) + l * 768, nullptr, vbuf, 768, 0, 0,
             (int)MS, 768, 768, 1.0f, 1, 1, 1);
        vtrans_kernel<<<(unsigned)((48ll * 64 * 512 + 255) / 256), 256, 0, stream>>>(
            vbuf, vt, 48ll * 64 * 512);

        // scores = Q * K^T  (48 batched 499x499x64)
        gemm(qb, 768, (long)S * 768, 64, kbuf, 768, (long)S * 768, 64, nullptr,
             scores, nullptr, 512, 12ll * S * 512, (long)S * 512,
             S, S, 64, 1.0f, 12, 0, 48);

        // softmax with gated relative-position bias
        softmax_kernel<<<48 * S, 256, 0, stream>>>(scores, gateb, posbias, probs, S, 512);

        // ctx = P * V  (48 batched 499x64x512, K zero-padded)
        gemm(probs, 512, 12ll * S * 512, (long)S * 512, vt, 512, 12ll * 64 * 512, 64ll * 512,
             nullptr, nullptr, ctxb, 768, (long)S * 768, 64,
             S, 64, 512, 1.0f, 12, 0, 48);

        // output projection + residual + LN1
        gemm(ctxb, 768, 0, 0, wo, 768, 0, 0, IN(21) + l * 768, attnf, nullptr, 768, 0, 0,
             (int)MS, 768, 768, 1.0f, 1, 1, 1);
        ln_rows<<<(unsigned)MS, 256, 0, stream>>>(attnf, xf, IN(25) + l * 768, IN(26) + l * 768,
                                                  xf, xb, 768, 0);

        // FFN
        gemm(xb, 768, 0, 0, w1, 768, 0, 0, IN(28) + l * 3072, nullptr, hb, 3072, 0, 0,
             (int)MS, 3072, 768, 1.0f, 1, 3, 1);
        gemm(hb, 3072, 0, 0, w2, 3072, 0, 0, IN(30) + l * 768, ff2f, nullptr, 768, 0, 0,
             (int)MS, 768, 3072, 1.0f, 1, 1, 1);
        ln_rows<<<(unsigned)MS, 256, 0, stream>>>(ff2f, xf, IN(31) + l * 768, IN(32) + l * 768,
                                                  xf, xb, 768, 0);
    }

    hipMemcpyAsync(d_out, xf, (size_t)MS * 768 * 4, hipMemcpyDeviceToDevice, stream);
}
